// Encoder_34694745817612
// MI455X (gfx1250) — matrix operand order repaired and checked by reading, not compiled, not run on hardware
//
#include <hip/hip_runtime.h>
#include <cstdint>
#include <cstddef>

// ---------------- shapes from the reference ----------------
constexpr int L_NUM   = 6;
constexpr int H_NUM   = 16;
constexpr int D_MODEL = 1024;
constexpr int D_K     = 64;
constexpr int D_BERT  = 768;
constexpr int F_DIM   = 4096;
constexpr int B_SZ    = 4;
constexpr int S_LEN   = 1024;
constexpr int MS      = B_SZ * S_LEN;   // 4096 token rows
constexpr int HDK     = H_NUM * D_K;    // 1024

typedef __bf16 bf16_t;
typedef bf16_t v16bf __attribute__((ext_vector_type(16)));
typedef float  v8f   __attribute__((ext_vector_type(8)));

struct __align__(16) U4 { unsigned a, b, c, d; };
union Frag   { v16bf v; U4 q[2]; };
union Pack8  { U4 u; unsigned short s[8]; };

__device__ __forceinline__ bf16_t f2bf(float f) { return (bf16_t)f; }  // HW RNE convert

__device__ __forceinline__ v8f wmma_bf16(const Frag& a, const Frag& b, v8f c) {
  // D = A(16x32 bf16) * B(32x16 bf16) + C(16x16 f32)
  return __builtin_amdgcn_wmma_f32_16x16x32_bf16(false, a.v, false, b.v,
                                                 (short)0, c, false, false);
}

// =====================================================================
// bf16 WMMA GEMM: C[M,N] = A[M,K] * B[K,N] + bias (+relu).
// Block tile 128x128, 8 waves (4x2), wave tile 32x64 (2x4 WMMA accums).
// Software-pipelined global->reg->LDS staging; global_prefetch 2 tiles ahead.
// OUTF=1 -> fp32 output; OUTF=0 -> bf16 output. (compile-time, branch-free)
// =====================================================================
template<int RELU, int OUTF>
__global__ __launch_bounds__(256) void gemm_bf16_wmma(
    const bf16_t* __restrict__ A, const bf16_t* __restrict__ Bm,
    const float* __restrict__ bias, void* __restrict__ Cout,
    int M, int N, int K)
{
  __shared__ __align__(16) bf16_t As[128][48];   // [m][k], padded row (96B)
  __shared__ __align__(16) bf16_t Bt[128][48];   // transposed: [n][k]

  const int tid  = threadIdx.x;
  const int lane = tid & 31, wv = tid >> 5;
  const int hf   = lane >> 4, l15 = lane & 15;
  const int wm   = wv >> 1,  wn  = wv & 1;
  const int m0   = blockIdx.y * 128, n0 = blockIdx.x * 128;
  const int wrow = wm * 32,  wcol = wn * 64;

  v8f acc[2][4];
  #pragma unroll
  for (int i = 0; i < 2; ++i)
    #pragma unroll
    for (int j = 0; j < 4; ++j)
      #pragma unroll
      for (int e = 0; e < 8; ++e) acc[i][j][e] = 0.f;

  // staging assignments
  const int ar = tid >> 1,        ac = (tid & 1) * 16;   // A: 2 thr/row, 16 elems
  const int bk = (tid >> 4) * 2,  bn = (tid & 15) * 8;   // B: k-pair x 8 n-elems

  const bf16_t* aptr = A  + (size_t)(m0 + ar) * K + ac;
  const bf16_t* bptr = Bm + (size_t)bk * N + n0 + bn;

  // prologue: tile 0 into registers
  U4 a0 = *(const U4*)(aptr);
  U4 a1 = *(const U4*)(aptr + 8);
  U4 b0 = *(const U4*)(bptr);
  U4 b1 = *(const U4*)(bptr + N);

  for (int k0 = 0; k0 < K; k0 += 32) {
    // stage registers into LDS
    *(U4*)&As[ar][ac]     = a0;
    *(U4*)&As[ar][ac + 8] = a1;
    {
      Pack8 r0, r1; r0.u = b0; r1.u = b1;
      #pragma unroll
      for (int i = 0; i < 8; ++i) {   // pack (k, k+1) pair -> one b32 store
        unsigned v = (unsigned)r0.s[i] | ((unsigned)r1.s[i] << 16);
        *(unsigned*)&Bt[bn + i][bk] = v;
      }
    }
    __syncthreads();

    // issue next tile's global loads; latency hidden behind frag loads + WMMA
    if (k0 + 32 < K) {
      a0 = *(const U4*)(aptr + k0 + 32);
      a1 = *(const U4*)(aptr + k0 + 40);
      b0 = *(const U4*)(bptr + (size_t)(k0 + 32) * N);
      b1 = *(const U4*)(bptr + (size_t)(k0 + 33) * N);
      if (k0 + 64 < K) {
        __builtin_prefetch((const void*)(aptr + k0 + 64), 0, 1);
        __builtin_prefetch((const void*)(bptr + (size_t)(k0 + 64) * N), 0, 1);
      }
    }

    Frag af[2], bfr[4];
    #pragma unroll
    for (int i = 0; i < 2; ++i) {   // A frag: lanes<16 k 0-7/16-23, lanes>=16 k 8-15/24-31
      const int r = wrow + i * 16 + l15;
      af[i].q[0] = *(const U4*)&As[r][hf * 8];
      af[i].q[1] = *(const U4*)&As[r][16 + hf * 8];
    }
    #pragma unroll
    for (int j = 0; j < 4; ++j) {   // B frag: lanes<16 k 0-7/16-23, lanes>=16 k 8-15/24-31 per N column
      const int n = wcol + j * 16 + l15;
      bfr[j].q[0] = *(const U4*)&Bt[n][hf * 8];
      bfr[j].q[1] = *(const U4*)&Bt[n][16 + hf * 8];
    }
    #pragma unroll
    for (int i = 0; i < 2; ++i)
      #pragma unroll
      for (int j = 0; j < 4; ++j)
        acc[i][j] = wmma_bf16(af[i], bfr[j], acc[i][j]);
    __syncthreads();
  }

  // branch-free epilogue: C layout => VGPR e: row = hf*8+e, col = l15
  #pragma unroll
  for (int i = 0; i < 2; ++i)
    #pragma unroll
    for (int j = 0; j < 4; ++j) {
      const int col = n0 + wcol + j * 16 + l15;
      const float bv = bias[col];
      #pragma unroll
      for (int e = 0; e < 8; ++e) {
        const int row = m0 + wrow + i * 16 + hf * 8 + e;
        float val = acc[i][j][e] + bv;
        if (RELU) val = fmaxf(val, 0.f);
        const size_t idx = (size_t)row * N + col;
        if (OUTF) ((float*)Cout)[idx] = val;
        else      ((bf16_t*)Cout)[idx] = f2bf(val);
      }
    }
}

// =====================================================================
// Fused flash-attention + per-head wh projection + bh bias.
// Transposed-score formulation: S^T = K_chunk(A) x Q^T(B), so softmax
// reductions are in-lane (+1 xor16) and P is produced directly in the
// A-fragment register layout for the P*V WMMA -- no P LDS staging.
// Grid: (S/64, B*H); block 128 (4 waves); one wave = 16 query rows.
// Q,K,V: bf16 [B*S, H*DK] (col = h*DK + dk). out: bf16 [B*S, H*DK].
// whT: bf16 [H][v][k] (pre-transposed), bh: f32 [H*DK].
// =====================================================================
__global__ __launch_bounds__(128) void attn_wmma(
    const bf16_t* __restrict__ Q, const bf16_t* __restrict__ Kmat,
    const bf16_t* __restrict__ Vmat, const unsigned char* __restrict__ mask,
    const bf16_t* __restrict__ whT, const float* __restrict__ bh,
    bf16_t* __restrict__ out)
{
  __shared__ __align__(16) bf16_t Ks[32][64];       // key chunk [key][dk]
  __shared__ __align__(16) bf16_t Vt[64][32];       // value chunk transposed [dk][key]
  __shared__ __align__(16) bf16_t Ost[4][16][64];   // per-wave O staging for wh

  const int tid  = threadIdx.x;
  const int lane = tid & 31, wv = tid >> 5;
  const int hf   = lane >> 4, l15 = lane & 15;
  const int b    = blockIdx.y >> 4;     // H = 16
  const int h    = blockIdx.y & 15;
  const int q0   = blockIdx.x * 64 + wv * 16;

  // Q as B-fragments (lane = query col, elems = contiguous dk) -- loaded once
  Frag bq[2];
  {
    const bf16_t* qp = Q + (size_t)(b * S_LEN + q0 + l15) * HDK + h * D_K;
    bq[0].q[0] = *(const U4*)(qp + hf * 16);
    bq[0].q[1] = *(const U4*)(qp + hf * 16 + 8);
    bq[1].q[0] = *(const U4*)(qp + 32 + hf * 16);
    bq[1].q[1] = *(const U4*)(qp + 32 + hf * 16 + 8);
  }

  // per-lane softmax state for query l15 (replicated in lanes l15 and l15+16)
  float mlane = -__builtin_inff(), llane = 0.f;
  v8f acc[4];
  #pragma unroll
  for (int j = 0; j < 4; ++j)
    #pragma unroll
    for (int e = 0; e < 8; ++e) acc[j][e] = 0.f;

  const unsigned char* mk = mask + (size_t)b * S_LEN;

  // staging assignments
  const int ld_key = tid >> 2, ld_c = (tid & 3) * 16;   // Ks: 4 thr/key
  const int kp = (tid >> 3) * 2, dc = (tid & 7) * 8;    // Vt: key-pair x 8 dk

  const bf16_t* kbase = Kmat + (size_t)(b * S_LEN) * HDK + h * D_K;
  const bf16_t* vbase = Vmat + (size_t)(b * S_LEN) * HDK + h * D_K;

  // prologue: chunk 0 into registers
  U4 kr0 = *(const U4*)(kbase + (size_t)ld_key * HDK + ld_c);
  U4 kr1 = *(const U4*)(kbase + (size_t)ld_key * HDK + ld_c + 8);
  U4 vr0 = *(const U4*)(vbase + (size_t)kp * HDK + dc);
  U4 vr1 = *(const U4*)(vbase + (size_t)(kp + 1) * HDK + dc);

  for (int t0 = 0; t0 < S_LEN; t0 += 32) {
    // stage current chunk into LDS
    *(U4*)&Ks[ld_key][ld_c]     = kr0;
    *(U4*)&Ks[ld_key][ld_c + 8] = kr1;
    {
      Pack8 r0, r1; r0.u = vr0; r1.u = vr1;
      #pragma unroll
      for (int i = 0; i < 8; ++i) {
        unsigned v = (unsigned)r0.s[i] | ((unsigned)r1.s[i] << 16);
        *(unsigned*)&Vt[dc + i][kp] = v;
      }
    }
    __syncthreads();

    // issue next chunk's global loads
    if (t0 + 32 < S_LEN) {
      kr0 = *(const U4*)(kbase + (size_t)(t0 + 32 + ld_key) * HDK + ld_c);
      kr1 = *(const U4*)(kbase + (size_t)(t0 + 32 + ld_key) * HDK + ld_c + 8);
      vr0 = *(const U4*)(vbase + (size_t)(t0 + 32 + kp) * HDK + dc);
      vr1 = *(const U4*)(vbase + (size_t)(t0 + 33 + kp) * HDK + dc);
    }

    // S^T tiles: M = keys (A = K chunk), N = queries (B = Q). Per tile:
    //   lane: query = l15, key = t*16 + hf*8 + e
    v8f sc[2];
    #pragma unroll
    for (int t = 0; t < 2; ++t) {
      const int key = t * 16 + l15;
      Frag ak0, ak1;
      ak0.q[0] = *(const U4*)&Ks[key][hf * 16];
      ak0.q[1] = *(const U4*)&Ks[key][hf * 16 + 8];
      ak1.q[0] = *(const U4*)&Ks[key][32 + hf * 16];
      ak1.q[1] = *(const U4*)&Ks[key][32 + hf * 16 + 8];
      v8f c;
      #pragma unroll
      for (int e = 0; e < 8; ++e) c[e] = 0.f;
      c = wmma_bf16(ak0, bq[0], c);
      c = wmma_bf16(ak1, bq[1], c);
      sc[t] = c;
    }

    // mask + scale; in-lane chunk max then one xor16 combine
    const unsigned long long mb0 = *(const unsigned long long*)(mk + t0 + hf * 8);
    const unsigned long long mb1 = *(const unsigned long long*)(mk + t0 + 16 + hf * 8);
    const float NEG = -__builtin_inff();
    float sv[2][8];
    float cm = NEG;
    #pragma unroll
    for (int e = 0; e < 8; ++e) {
      const bool k0m = ((mb0 >> (8 * e)) & 0xffull) != 0;
      const bool k1m = ((mb1 >> (8 * e)) & 0xffull) != 0;
      sv[0][e] = k0m ? NEG : sc[0][e] * 0.125f;   // DK^-0.5
      sv[1][e] = k1m ? NEG : sc[1][e] * 0.125f;
      cm = fmaxf(cm, fmaxf(sv[0][e], sv[1][e]));
    }
    cm = fmaxf(cm, __shfl_xor(cm, 16, 32));

    const float mn  = fmaxf(mlane, cm);
    const float mc  = fmaxf(mn, -1e30f);          // clamp: NaN-free when all masked
    const float scl = __expf(mlane - mc);
    mlane = mn;

    // P = exp(s - mc), packed straight into the A-fragment register layout:
    // elems 0-7 = tile0 keys, elems 8-15 = tile1 keys (half-wave split matches ISA)
    Frag ap;
    float rs = 0.f;
    #pragma unroll
    for (int e = 0; e < 8; ++e) {
      const float p0 = __expf(sv[0][e] - mc);
      const float p1 = __expf(sv[1][e] - mc);
      rs += p0 + p1;
      ap.v[e]     = f2bf(p0);
      ap.v[8 + e] = f2bf(p1);
    }
    rs += __shfl_xor(rs, 16, 32);
    llane = llane * scl + rs;

    // broadcast rescale factor into O-accumulator C-layout rows (8 bpermutes)
    float srow[8];
    #pragma unroll
    for (int e = 0; e < 8; ++e) srow[e] = __shfl(scl, hf * 8 + e, 32);
    #pragma unroll
    for (int j = 0; j < 4; ++j)
      #pragma unroll
      for (int e = 0; e < 8; ++e) acc[j][e] *= srow[e];

    // O += P * V (4 WMMAs, B-frags from transposed V in LDS)
    #pragma unroll
    for (int j = 0; j < 4; ++j) {
      const int n = j * 16 + l15;
      Frag bvf;
      bvf.q[0] = *(const U4*)&Vt[n][hf * 8];
      bvf.q[1] = *(const U4*)&Vt[n][16 + hf * 8];
      acc[j] = wmma_bf16(ap, bvf, acc[j]);
    }
    __syncthreads();
  }

  // epilogue: normalize (broadcast 1/l into C-layout), then fused wh + bh
  const float invl = 1.f / llane;
  float irow[8];
  #pragma unroll
  for (int e = 0; e < 8; ++e) irow[e] = __shfl(invl, hf * 8 + e, 32);
  #pragma unroll
  for (int e = 0; e < 8; ++e)
    #pragma unroll
    for (int j = 0; j < 4; ++j)
      Ost[wv][hf * 8 + e][j * 16 + l15] = f2bf(acc[j][e] * irow[e]);

  Frag ao[2];
  ao[0].q[0] = *(const U4*)&Ost[wv][l15][hf * 8];
  ao[0].q[1] = *(const U4*)&Ost[wv][l15][16 + hf * 8];
  ao[1].q[0] = *(const U4*)&Ost[wv][l15][32 + hf * 8];
  ao[1].q[1] = *(const U4*)&Ost[wv][l15][48 + hf * 8];
  #pragma unroll
  for (int j = 0; j < 4; ++j) {
    const int vo = j * 16 + l15;
    const bf16_t* wp = whT + ((size_t)h * D_K + vo) * D_K;
    Frag w0, w1;
    w0.q[0] = *(const U4*)(wp + hf * 8);
    w0.q[1] = *(const U4*)(wp + 16 + hf * 8);
    w1.q[0] = *(const U4*)(wp + 32 + hf * 8);
    w1.q[1] = *(const U4*)(wp + 48 + hf * 8);
    v8f r;
    #pragma unroll
    for (int e = 0; e < 8; ++e) r[e] = 0.f;
    r = wmma_bf16(ao[0], w0, r);
    r = wmma_bf16(ao[1], w1, r);
    const float bb = bh[h * D_K + vo];
    #pragma unroll
    for (int e = 0; e < 8; ++e) {
      const int row = q0 + hf * 8 + e;
      out[(size_t)(b * S_LEN + row) * HDK + h * D_K + vo] = f2bf(r[e] + bb);
    }
  }
}

// =====================================================================
// Fused residual-combine + LayerNorm over D=1024; one block per row.
// NIN=3: t = ca*a + cb*b + cc*c ; NIN=2: t = ca*a + cb*b. Branch-free.
// =====================================================================
template<int NIN>
__global__ __launch_bounds__(256) void ln_kernel(
    const float* __restrict__ a, const float* __restrict__ b, const float* __restrict__ c,
    float ca, float cb, float cc,
    const float* __restrict__ g, const float* __restrict__ beta,
    float* __restrict__ outf, bf16_t* __restrict__ outb)
{
  __shared__ float red[2][8];
  const int row = blockIdx.x, tid = threadIdx.x;
  const size_t base = (size_t)row * D_MODEL;
  float t[4], s = 0.f, s2 = 0.f;
  #pragma unroll
  for (int i = 0; i < 4; ++i) {
    const int idx = tid + i * 256;
    float v = ca * a[base + idx] + cb * b[base + idx];
    if (NIN >= 3) v += cc * c[base + idx];
    t[i] = v; s += v; s2 += v * v;
  }
  #pragma unroll
  for (int sh = 1; sh < 32; sh <<= 1) { s += __shfl_xor(s, sh, 32); s2 += __shfl_xor(s2, sh, 32); }
  if ((tid & 31) == 0) { red[0][tid >> 5] = s; red[1][tid >> 5] = s2; }
  __syncthreads();
  float ts = 0.f, ts2 = 0.f;
  #pragma unroll
  for (int i = 0; i < 8; ++i) { ts += red[0][i]; ts2 += red[1][i]; }
  const float mean = ts * (1.f / D_MODEL);
  const float var  = ts2 * (1.f / D_MODEL) - mean * mean;
  const float rstd = rsqrtf(var + 1e-5f);
  #pragma unroll
  for (int i = 0; i < 4; ++i) {
    const int idx = tid + i * 256;
    const float o = (t[i] - mean) * rstd * g[idx] + beta[idx];
    outf[base + idx] = o;
    outb[base + idx] = f2bf(o);
  }
}

// ----------------- small convert / permute kernels -----------------
__global__ void cvt_bf16_kernel(const float* __restrict__ in, bf16_t* __restrict__ out, int n) {
  int i = blockIdx.x * 256 + threadIdx.x;
  if (i < n) out[i] = f2bf(in[i]);
}
__global__ void cvt_pair_kernel(const float* __restrict__ in, float* __restrict__ outf,
                                bf16_t* __restrict__ outb, int n) {
  int i = blockIdx.x * 256 + threadIdx.x;
  if (i < n) { float v = in[i]; outf[i] = v; outb[i] = f2bf(v); }
}
// wq[H][Din][DK] (f32) -> out[Din][H*DK] (bf16); HDK==1024 via shifts
__global__ void perm_qkv_kernel(const float* __restrict__ in, bf16_t* __restrict__ out, int Din) {
  int o = blockIdx.x * 256 + threadIdx.x;
  if (o >= Din * HDK) return;
  int d = o >> 10, rest = o & 1023, h = rest >> 6, k = rest & 63;
  out[o] = f2bf(in[((size_t)h * Din + d) * D_K + k]);
}
// wh[H][k][v] (f32) -> whT[H][v][k] (bf16)
__global__ void perm_wh_kernel(const float* __restrict__ in, bf16_t* __restrict__ out) {
  int i = blockIdx.x * 256 + threadIdx.x;
  if (i >= H_NUM * D_K * D_K) return;
  int h = i >> 12, v = (i >> 6) & 63, k = i & 63;
  out[i] = f2bf(in[(((size_t)h << 6) + k) * 64 + v]);
}

// =====================================================================
extern "C" void kernel_launch(void* const* d_in, const int* in_sizes, int n_in,
                              void* d_out, int out_size, void* d_ws, size_t ws_size,
                              hipStream_t stream) {
  (void)in_sizes; (void)n_in; (void)out_size; (void)ws_size;
  const float* x_in  = (const float*)d_in[0];
  const float* bert  = (const float*)d_in[1];
  const unsigned char* mask = (const unsigned char*)d_in[2];
  const float* s_wq = (const float*)d_in[3];
  const float* s_bq = (const float*)d_in[4];
  const float* s_wk = (const float*)d_in[5];
  const float* s_bk = (const float*)d_in[6];
  const float* s_wv = (const float*)d_in[7];
  const float* s_bv = (const float*)d_in[8];
  const float* s_wh = (const float*)d_in[9];
  const float* s_bh = (const float*)d_in[10];
  const float* s_wo = (const float*)d_in[11];
  const float* s_bo = (const float*)d_in[12];
  const float* b_wq = (const float*)d_in[13];
  const float* b_bq = (const float*)d_in[14];
  const float* b_wk = (const float*)d_in[15];
  const float* b_bk = (const float*)d_in[16];
  const float* b_wv = (const float*)d_in[17];
  const float* b_bv = (const float*)d_in[18];
  const float* b_wh = (const float*)d_in[19];
  const float* b_bh = (const float*)d_in[20];
  const float* b_wo = (const float*)d_in[21];
  const float* b_bo = (const float*)d_in[22];
  const float* w1   = (const float*)d_in[23];
  const float* b1   = (const float*)d_in[24];
  const float* w2   = (const float*)d_in[25];
  const float* b2   = (const float*)d_in[26];
  const float* ln1_g = (const float*)d_in[27];
  const float* ln1_b = (const float*)d_in[28];
  const float* ln2_g = (const float*)d_in[29];
  const float* ln2_b = (const float*)d_in[30];

  // ---- carve workspace ----
  char* p = (char*)d_ws;
  auto alloc = [&](size_t bytes) { char* r = p; p += (bytes + 255) & ~(size_t)255; return (void*)r; };
  float*  xf    = (float*) alloc((size_t)MS * D_MODEL * 4);
  bf16_t* xb    = (bf16_t*)alloc((size_t)MS * D_MODEL * 2);
  bf16_t* bertb = (bf16_t*)alloc((size_t)MS * D_BERT * 2);
  bf16_t* qb    = (bf16_t*)alloc((size_t)MS * HDK * 2);
  bf16_t* kb    = (bf16_t*)alloc((size_t)MS * HDK * 2);
  bf16_t* vb    = (bf16_t*)alloc((size_t)MS * HDK * 2);
  bf16_t* ctxb  = (bf16_t*)alloc((size_t)MS * HDK * 2);
  bf16_t* whT   = (bf16_t*)alloc((size_t)H_NUM * D_K * D_K * 2);
  float*  sa    = (float*) alloc((size_t)MS * D_MODEL * 4);
  float*  ba    = (float*) alloc((size_t)MS * D_MODEL * 4);
  float*  yf    = (float*) alloc((size_t)MS * D_MODEL * 4);
  bf16_t* yb    = (bf16_t*)alloc((size_t)MS * D_MODEL * 2);
  bf16_t* h1b   = (bf16_t*)alloc((size_t)MS * F_DIM * 2);
  float*  ffb   = (float*) alloc((size_t)MS * D_MODEL * 4);
  bf16_t* wbuf  = (bf16_t*)alloc((size_t)D_MODEL * F_DIM * 2);   // largest weight

  // specialized GEMM launchers (bias always present)
  auto gemmB = [&](const bf16_t* A, const bf16_t* Bm, const float* bias,
                   bf16_t* C, int M, int N, int K) {   // bf16 out
    dim3 g(N / 128, M / 128);
    gemm_bf16_wmma<0, 0><<<g, 256, 0, stream>>>(A, Bm, bias, (void*)C, M, N, K);
  };
  auto gemmBR = [&](const bf16_t* A, const bf16_t* Bm, const float* bias,
                    bf16_t* C, int M, int N, int K) {  // bf16 out + relu
    dim3 g(N / 128, M / 128);
    gemm_bf16_wmma<1, 0><<<g, 256, 0, stream>>>(A, Bm, bias, (void*)C, M, N, K);
  };
  auto gemmF = [&](const bf16_t* A, const bf16_t* Bm, const float* bias,
                   float* C, int M, int N, int K) {    // fp32 out
    dim3 g(N / 128, M / 128);
    gemm_bf16_wmma<0, 1><<<g, 256, 0, stream>>>(A, Bm, bias, (void*)C, M, N, K);
  };
  auto cvt = [&](const float* in, bf16_t* out, size_t n) {
    cvt_bf16_kernel<<<dim3((unsigned)((n + 255) / 256)), 256, 0, stream>>>(in, out, (int)n);
  };
  auto perm = [&](const float* in, bf16_t* out, int Din) {
    size_t n = (size_t)Din * HDK;
    perm_qkv_kernel<<<dim3((unsigned)((n + 255) / 256)), 256, 0, stream>>>(in, out, Din);
  };

  // layer strides
  const size_t sW   = (size_t)H_NUM * D_MODEL * D_K;   // s_wq/s_wk/s_wv/b_wq
  const size_t sWb  = (size_t)H_NUM * D_BERT * D_K;    // b_wk/b_wv
  const size_t sWh  = (size_t)H_NUM * D_K * D_K;
  const size_t sWo  = (size_t)D_MODEL * D_MODEL;
  const size_t sW12 = (size_t)D_MODEL * F_DIM;

  cvt_pair_kernel<<<dim3((MS * D_MODEL + 255) / 256), 256, 0, stream>>>(x_in, xf, xb, MS * D_MODEL);
  cvt(bert, bertb, (size_t)MS * D_BERT);

  for (int l = 0; l < L_NUM; ++l) {
    // ---------------- self attention ----------------
    perm(s_wq + l * sW, wbuf, D_MODEL);
    gemmB(xb, wbuf, s_bq + l * HDK, qb, MS, HDK, D_MODEL);
    perm(s_wk + l * sW, wbuf, D_MODEL);
    gemmB(xb, wbuf, s_bk + l * HDK, kb, MS, HDK, D_MODEL);
    perm(s_wv + l * sW, wbuf, D_MODEL);
    gemmB(xb, wbuf, s_bv + l * HDK, vb, MS, HDK, D_MODEL);
    perm_wh_kernel<<<dim3((H_NUM * D_K * D_K + 255) / 256), 256, 0, stream>>>(s_wh + l * sWh, whT);
    attn_wmma<<<dim3(S_LEN / 64, B_SZ * H_NUM), 128, 0, stream>>>(
        qb, kb, vb, mask, whT, s_bh + l * HDK, ctxb);
    cvt(s_wo + l * sWo, wbuf, sWo);
    gemmF(ctxb, wbuf, s_bo + l * D_MODEL, sa, MS, D_MODEL, D_MODEL);

    // ---------------- cross attention (K/V from BERT embedding) ----------------
    perm(b_wq + l * sW, wbuf, D_MODEL);
    gemmB(xb, wbuf, b_bq + l * HDK, qb, MS, HDK, D_MODEL);
    perm(b_wk + l * sWb, wbuf, D_BERT);
    gemmB(bertb, wbuf, b_bk + l * HDK, kb, MS, HDK, D_BERT);
    perm(b_wv + l * sWb, wbuf, D_BERT);
    gemmB(bertb, wbuf, b_bv + l * HDK, vb, MS, HDK, D_BERT);
    perm_wh_kernel<<<dim3((H_NUM * D_K * D_K + 255) / 256), 256, 0, stream>>>(b_wh + l * sWh, whT);
    attn_wmma<<<dim3(S_LEN / 64, B_SZ * H_NUM), 128, 0, stream>>>(
        qb, kb, vb, mask, whT, b_bh + l * HDK, ctxb);
    cvt(b_wo + l * sWo, wbuf, sWo);
    gemmF(ctxb, wbuf, b_bo + l * D_MODEL, ba, MS, D_MODEL, D_MODEL);

    // ---------------- LN1: y = LN(0.5*sa + 0.5*ba + x) ----------------
    ln_kernel<3><<<dim3(MS), 256, 0, stream>>>(sa, ba, xf, 0.5f, 0.5f, 1.0f,
                                               ln1_g + l * D_MODEL, ln1_b + l * D_MODEL, yf, yb);

    // ---------------- FFN ----------------
    cvt(w1 + l * sW12, wbuf, sW12);
    gemmBR(yb, wbuf, b1 + l * F_DIM, h1b, MS, F_DIM, D_MODEL);   // ReLU fused
    cvt(w2 + l * sW12, wbuf, sW12);
    gemmF(h1b, wbuf, b2 + l * D_MODEL, ffb, MS, D_MODEL, F_DIM);

    // ---------------- LN2: x = LN(ff + y) ----------------
    float* xo = (l == L_NUM - 1) ? (float*)d_out : xf;
    ln_kernel<2><<<dim3(MS), 256, 0, stream>>>(ffb, yf, nullptr, 1.f, 1.f, 0.f,
                                               ln2_g + l * D_MODEL, ln2_b + l * D_MODEL, xo, xb);
  }
}
